// BareMambaLayer_51539607552222
// MI455X (gfx1250) — matrix operand order repaired and checked by reading, not compiled, not run on hardware
//
#include <hip/hip_runtime.h>
#include <hip/hip_bf16.h>
#include <math.h>

// ---------------------------------------------------------------------------
// Problem constants (from reference setup_inputs)
// b=4, l=4096, d_model=1024, expand=2, d_state=128, headdim=64
// d_inner=2048, nheads=32, conv_dim=2304, d_in_proj=4384
// ---------------------------------------------------------------------------
#define BB     4
#define LL     4096
#define DM     1024
#define DI     2048
#define NH     32
#define HD     64
#define DSTATE 128
#define CDIM   2304
#define DPROJ  4384
#define MM     (BB * LL)          // 16384 rows
#define CHQ    64                 // chunk length
#define NCH    (LL / CHQ)         // 64 chunks per batch
#define NBHC   (BB * NH * NCH)    // 8192 chunk workgroups

// CDNA5 async global->LDS staging path (ASYNCcnt).
#define USE_ASYNC_LDS 1

typedef __bf16 bf16;
typedef __attribute__((ext_vector_type(16))) __bf16 v16bf;
typedef __attribute__((ext_vector_type(8)))  __bf16 v8bf;
typedef __attribute__((ext_vector_type(8)))  float  v8f;

static __device__ __forceinline__ bf16 to_bf16(float f) {
    union { float f; unsigned int u; } in; in.f = f;
    unsigned int u = in.u + 0x7FFFu + ((in.u >> 16) & 1u);   // round-to-nearest-even
    union { unsigned short s; bf16 b; } out;
    out.s = (unsigned short)(u >> 16);
    return out.b;
}
static __device__ __forceinline__ float bf16_to_f32(bf16 b) {
    union { bf16 b; unsigned short s; } in; in.b = b;
    union { unsigned int u; float f; } out; out.u = ((unsigned)in.s) << 16;
    return out.f;
}
static __device__ __forceinline__ float silu(float x) {
    return x / (1.f + __expf(-x));
}
static __device__ __forceinline__ v8f wmma_bf16(v16bf a, v16bf b, v8f c) {
    return __builtin_amdgcn_wmma_f32_16x16x32_bf16(false, a, false, b, (short)0, c,
                                                   false, false);
}

#if USE_ASYNC_LDS
static __device__ __forceinline__ void g2l_async_b128(unsigned lds, const void* g) {
    asm volatile("global_load_async_to_lds_b128 %0, %1, off"
                 :: "v"(lds), "v"(g) : "memory");
}
static __device__ __forceinline__ void wait_async0() {
    asm volatile("s_wait_asynccnt 0x0" ::: "memory");
}
static __device__ __forceinline__ unsigned lds_off(const void* p) {
    return (unsigned)(size_t)p;   // low 32 bits of flat LDS pointer = LDS offset
}
#endif

// ---------------------------------------------------------------------------
// fp32 -> bf16 elementwise conversion
// ---------------------------------------------------------------------------
__global__ __launch_bounds__(256) void k_f32_to_bf16(const float* __restrict__ in,
                                                     bf16* __restrict__ out, size_t n) {
    size_t i = (size_t)blockIdx.x * 256 + threadIdx.x;
    if (i < n) out[i] = to_bf16(in[i]);
}

// ---------------------------------------------------------------------------
// Tiled WMMA GEMM: C[M,N] = A[M,K] * W[N,K]^T   (A,W bf16 row-major, C fp32)
// Block tile 128x128, double-buffered LDS, async global->LDS staging.
// ---------------------------------------------------------------------------
__global__ __launch_bounds__(256) void k_gemm_bf16_wmma(const bf16* __restrict__ A,
                                                        const bf16* __restrict__ W,
                                                        float* __restrict__ C,
                                                        int M, int N, int K) {
    __shared__ alignas(32) bf16 sA[2][128][48];
    __shared__ alignas(32) bf16 sW[2][128][48];

    const int t    = threadIdx.x;
    const int lane = t & 31;
    const int wave = t >> 5;
    const int wr   = wave & 3;
    const int wc   = wave >> 2;
    const int lrow = lane & 15;
    const int g    = lane >> 4;

    const int mBase = blockIdx.y * 128;
    const int nBase = blockIdx.x * 128;

    const int row = t >> 1;
    const int seg = (t & 1) * 16;

    const bf16* gArow = A + (size_t)(mBase + row) * K + seg;
    int wn = nBase + row;
    if (wn > N - 1) wn = N - 1;      // branch-free OOB (cols >= N never stored)
    const bf16* gWrow = W + (size_t)wn * K + seg;

    v8f acc[2][4];
    #pragma unroll
    for (int mi = 0; mi < 2; ++mi)
        #pragma unroll
        for (int ni = 0; ni < 4; ++ni)
            acc[mi][ni] = (v8f)(0.f);

#if USE_ASYNC_LDS
    {
        const unsigned lA = lds_off(&sA[0][row][seg]);
        const unsigned lW = lds_off(&sW[0][row][seg]);
        g2l_async_b128(lA,      gArow);
        g2l_async_b128(lA + 16, gArow + 8);
        g2l_async_b128(lW,      gWrow);
        g2l_async_b128(lW + 16, gWrow + 8);
    }
    wait_async0();
    __syncthreads();
#else
    {
        const uint4* ga = (const uint4*)gArow;
        const uint4* gw = (const uint4*)gWrow;
        uint4 a0 = ga[0], a1 = ga[1], w0 = gw[0], w1 = gw[1];
        *(uint4*)&sA[0][row][seg]     = a0;
        *(uint4*)&sA[0][row][seg + 8] = a1;
        *(uint4*)&sW[0][row][seg]     = w0;
        *(uint4*)&sW[0][row][seg + 8] = w1;
    }
    __syncthreads();
#endif

    for (int k0 = 0; k0 < K; k0 += 32) {
        const int  cur  = (k0 >> 5) & 1;
        const int  nxt  = cur ^ 1;
        const bool more = (k0 + 32) < K;

#if USE_ASYNC_LDS
        if (more) {
            const unsigned lA = lds_off(&sA[nxt][row][seg]);
            const unsigned lW = lds_off(&sW[nxt][row][seg]);
            const bf16* ga = gArow + k0 + 32;
            const bf16* gw = gWrow + k0 + 32;
            g2l_async_b128(lA,      ga);
            g2l_async_b128(lA + 16, ga + 8);
            g2l_async_b128(lW,      gw);
            g2l_async_b128(lW + 16, gw + 8);
        }
#else
        uint4 a0, a1, w0, w1;
        if (more) {
            const uint4* ga = (const uint4*)(gArow + k0 + 32);
            const uint4* gw = (const uint4*)(gWrow + k0 + 32);
            a0 = ga[0]; a1 = ga[1]; w0 = gw[0]; w1 = gw[1];
        }
#endif

        union { v16bf v; v8bf h[2]; } af[2];
        #pragma unroll
        for (int mi = 0; mi < 2; ++mi) {
            const int r = wr * 32 + mi * 16 + lrow;
            af[mi].h[0] = *(const v8bf*)&sA[cur][r][16 * g];
            af[mi].h[1] = *(const v8bf*)&sA[cur][r][16 * g + 8];
        }
        v16bf bfr[4];
        #pragma unroll
        for (int ni = 0; ni < 4; ++ni) {
            const int n = wc * 64 + ni * 16 + lrow;
            bfr[ni] = *(const v16bf*)&sW[cur][n][16 * g];
        }
        #pragma unroll
        for (int mi = 0; mi < 2; ++mi)
            #pragma unroll
            for (int ni = 0; ni < 4; ++ni)
                acc[mi][ni] = wmma_bf16(af[mi].v, bfr[ni], acc[mi][ni]);

#if USE_ASYNC_LDS
        wait_async0();
        __syncthreads();
#else
        if (more) {
            *(uint4*)&sA[nxt][row][seg]     = a0;
            *(uint4*)&sA[nxt][row][seg + 8] = a1;
            *(uint4*)&sW[nxt][row][seg]     = w0;
            *(uint4*)&sW[nxt][row][seg + 8] = w1;
        }
        __syncthreads();
#endif
    }

    #pragma unroll
    for (int mi = 0; mi < 2; ++mi) {
        #pragma unroll
        for (int ni = 0; ni < 4; ++ni) {
            const int col = nBase + wc * 64 + ni * 16 + lrow;
            if (col < N) {
                union { v8f v; float f[8]; } res;
                res.v = acc[mi][ni];
                #pragma unroll
                for (int i = 0; i < 8; ++i) {
                    const int rowi = mBase + wr * 32 + mi * 16 + g * 8 + i;
                    C[(size_t)rowi * N + col] = res.f[i];
                }
            }
        }
    }
}

// ---------------------------------------------------------------------------
// Depthwise causal conv (4-tap) + bias + SiLU over the xBC slice of zxbcdt
// ---------------------------------------------------------------------------
__global__ __launch_bounds__(256) void k_conv_silu(const float* __restrict__ zx,
                                                   const float* __restrict__ cw,
                                                   const float* __restrict__ cb,
                                                   float* __restrict__ xBCc) {
    const size_t total = (size_t)MM * CDIM;
    size_t idx = (size_t)blockIdx.x * 256 + threadIdx.x;
    if (idx >= total) return;
    const int    c  = (int)(idx % CDIM);
    const size_t bl = idx / CDIM;
    const int    l  = (int)(bl & (LL - 1));

    float acc = cb[c];
    #pragma unroll
    for (int k = 0; k < 4; ++k) {
        const int ls = l - 3 + k;
        if (ls >= 0)
            acc += cw[c * 4 + k] * zx[(bl - 3 + k) * DPROJ + DI + c];
    }
    xBCc[idx] = silu(acc);
}

// ---------------------------------------------------------------------------
// dt = softplus(dt_raw + dt_bias); loga = -exp(A_log) * dt   (log of decay)
// ---------------------------------------------------------------------------
__global__ __launch_bounds__(256) void k_dt_decay(const float* __restrict__ zx,
                                                  const float* __restrict__ dt_bias,
                                                  const float* __restrict__ A_log,
                                                  float* __restrict__ dtv,
                                                  float* __restrict__ logav) {
    const size_t total = (size_t)MM * NH;
    size_t idx = (size_t)blockIdx.x * 256 + threadIdx.x;
    if (idx >= total) return;
    const int    h   = (int)(idx & (NH - 1));
    const size_t row = idx >> 5;
    const float raw = zx[row * DPROJ + (DI + CDIM) + h] + dt_bias[h];
    const float dt  = (raw > 20.f) ? raw : log1pf(__expf(raw));
    dtv[idx]   = dt;
    logav[idx] = -__expf(A_log[h]) * dt;
}

// ---------------------------------------------------------------------------
// Chunked scan, phase 1 (per b,h,chunk): chunk state via WMMA.
//   S_c[n][p] = sum_j exp(La_last - La_j) * dt_j * B_j[n] * x_j[p]
//   A_c      = exp(La_last)
// GEMM (128 x 64, K=64): A-op = scaled-B^T staged in LDS, B-op = X^T.
// ---------------------------------------------------------------------------
__global__ __launch_bounds__(256) void k_chunk_state(const float* __restrict__ xBCc,
                                                     const float* __restrict__ dtv,
                                                     const float* __restrict__ logav,
                                                     float* __restrict__ chunkS,
                                                     float* __restrict__ chunkA) {
    __shared__ alignas(32) bf16 sBt[DSTATE][CHQ];   // [n][j] scaled
    __shared__ alignas(32) bf16 sXt[HD][CHQ];       // [p][j]
    __shared__ float sdt[CHQ], sraw[CHQ], sLa[CHQ];

    const int bid = blockIdx.x;
    const int c   = bid & (NCH - 1);
    const int h   = (bid >> 6) & (NH - 1);
    const int b   = bid >> 11;
    const int bh  = b * NH + h;
    const int t   = threadIdx.x;
    const size_t rowBase = (size_t)b * LL + (size_t)c * CHQ;

    if (t < CHQ) {
        const size_t r = rowBase + t;
        sdt[t]  = dtv[r * NH + h];
        sraw[t] = logav[r * NH + h];
    }
    __syncthreads();
    if (t == 0) {           // serial prefix sum of log-decay (64 adds, negligible)
        float s = 0.f;
        for (int i = 0; i < CHQ; ++i) { s += sraw[i]; sLa[i] = s; }
    }
    __syncthreads();
    const float laLast = sLa[CHQ - 1];

    for (int idx = t; idx < DSTATE * CHQ; idx += 256) {
        const int n = idx >> 6, j = idx & (CHQ - 1);
        const float scale = __expf(laLast - sLa[j]) * sdt[j];
        sBt[n][j] = to_bf16(scale * xBCc[(rowBase + j) * CDIM + DI + n]);
    }
    for (int idx = t; idx < HD * CHQ; idx += 256) {
        const int p = idx >> 6, j = idx & (CHQ - 1);
        sXt[p][j] = to_bf16(xBCc[(rowBase + j) * CDIM + h * HD + p]);
    }
    __syncthreads();

    const int lane = t & 31, wave = t >> 5;
    const int lrow = lane & 15, g = lane >> 4;

    v8f acc[4];
    #pragma unroll
    for (int ni = 0; ni < 4; ++ni) acc[ni] = (v8f)(0.f);

    #pragma unroll
    for (int k0 = 0; k0 < CHQ; k0 += 32) {
        union { v16bf v; v8bf hh[2]; } af;
        af.hh[0] = *(const v8bf*)&sBt[wave * 16 + lrow][k0 + 16 * g];
        af.hh[1] = *(const v8bf*)&sBt[wave * 16 + lrow][k0 + 16 * g + 8];
        #pragma unroll
        for (int ni = 0; ni < 4; ++ni) {
            v16bf bb = *(const v16bf*)&sXt[ni * 16 + lrow][k0 + 16 * g];
            acc[ni] = wmma_bf16(af.v, bb, acc[ni]);
        }
    }

    const size_t base = ((size_t)bh * NCH + c) * (DSTATE * HD);
    #pragma unroll
    for (int ni = 0; ni < 4; ++ni) {
        union { v8f v; float f[8]; } res; res.v = acc[ni];
        const int p = ni * 16 + lrow;
        #pragma unroll
        for (int e = 0; e < 8; ++e) {
            const int n = wave * 16 + g * 8 + e;
            chunkS[base + (size_t)n * HD + p] = res.f[e];
        }
    }
    if (t == 0) chunkA[(size_t)bh * NCH + c] = __expf(laLast);
}

// ---------------------------------------------------------------------------
// Chunked scan, phase 2 (per b,h): 64-step inter-chunk recurrence.
//   Hprev(c) = H ; H = A_c * H + S_c   (8192 floats, 32 per thread, float4)
// ---------------------------------------------------------------------------
__global__ __launch_bounds__(256) void k_chunk_scan(const float* __restrict__ chunkS,
                                                    const float* __restrict__ chunkA,
                                                    float* __restrict__ chunkH) {
    const int bh = blockIdx.x;
    const int t  = threadIdx.x;
    const int off = t * 32;

    float4 H[8];
    #pragma unroll
    for (int j = 0; j < 8; ++j) { H[j].x = H[j].y = H[j].z = H[j].w = 0.f; }

    for (int c = 0; c < NCH; ++c) {
        const size_t base = ((size_t)bh * NCH + c) * (DSTATE * HD);
        float4*       dst = (float4*)(chunkH + base + off);
        const float4* src = (const float4*)(chunkS + base + off);
        const float Ac = chunkA[(size_t)bh * NCH + c];
        #pragma unroll
        for (int j = 0; j < 8; ++j) {
            dst[j] = H[j];                       // state BEFORE chunk c
            const float4 s = src[j];
            H[j].x = Ac * H[j].x + s.x;
            H[j].y = Ac * H[j].y + s.y;
            H[j].z = Ac * H[j].z + s.z;
            H[j].w = Ac * H[j].w + s.w;
        }
    }
}

// ---------------------------------------------------------------------------
// Chunked scan, phase 3 (per b,h,chunk): outputs via 3 WMMA GEMMs.
//   CB = C * B^T (64x64,K=128); G = mask/scale(CB);
//   Y  = G * X (K=64)  +  diag(exp(La)) * (C * Hprev) (K=128)  +  D*x
// ---------------------------------------------------------------------------
__global__ __launch_bounds__(256) void k_chunk_out(const float* __restrict__ xBCc,
                                                   const float* __restrict__ dtv,
                                                   const float* __restrict__ logav,
                                                   const float* __restrict__ chunkH,
                                                   const float* __restrict__ Dp,
                                                   float* __restrict__ y) {
    __shared__ alignas(32) bf16 sC[CHQ][DSTATE];
    __shared__ alignas(32) bf16 sB[CHQ][DSTATE];
    __shared__ alignas(32) bf16 sXt[HD][CHQ];       // [p][i]
    __shared__ alignas(32) bf16 sHt[HD][DSTATE];    // [p][n]
    __shared__ alignas(32) bf16 sG[CHQ][CHQ];
    __shared__ float sdt[CHQ], sraw[CHQ], sLa[CHQ];

    const int bid = blockIdx.x;
    const int c   = bid & (NCH - 1);
    const int h   = (bid >> 6) & (NH - 1);
    const int b   = bid >> 11;
    const int bh  = b * NH + h;
    const int t   = threadIdx.x;
    const size_t rowBase = (size_t)b * LL + (size_t)c * CHQ;

    // ---- stage chunk operands ----
    for (int idx = t; idx < CHQ * DSTATE; idx += 256) {
        const int i = idx >> 7, n = idx & (DSTATE - 1);
        const float* xrow = xBCc + (rowBase + i) * CDIM;
        sB[i][n] = to_bf16(xrow[DI + n]);
        sC[i][n] = to_bf16(xrow[DI + DSTATE + n]);
    }
    for (int idx = t; idx < HD * CHQ; idx += 256) {
        const int p = idx >> 6, i = idx & (CHQ - 1);
        sXt[p][i] = to_bf16(xBCc[(rowBase + i) * CDIM + h * HD + p]);
    }
    const float* Hp = chunkH + ((size_t)bh * NCH + c) * (DSTATE * HD);
    for (int idx = t; idx < DSTATE * HD; idx += 256) {
        const int n = idx >> 6, p = idx & (HD - 1);
        sHt[p][n] = to_bf16(Hp[(size_t)n * HD + p]);
    }
    if (t < CHQ) {
        const size_t r = rowBase + t;
        sdt[t]  = dtv[r * NH + h];
        sraw[t] = logav[r * NH + h];
    }
    __syncthreads();
    if (t == 0) {
        float s = 0.f;
        for (int i = 0; i < CHQ; ++i) { s += sraw[i]; sLa[i] = s; }
    }
    __syncthreads();

    const int lane = t & 31, wave = t >> 5;
    const int lrow = lane & 15, g = lane >> 4;
    const int wr = wave & 3;        // row tile of 16 (i)
    const int wc = wave >> 2;       // col-tile pair (j or p)

    // ---- GEMM1: CB[i][j], K = 128 ----
    v8f accG[2];
    accG[0] = (v8f)(0.f); accG[1] = (v8f)(0.f);
    #pragma unroll
    for (int k0 = 0; k0 < DSTATE; k0 += 32) {
        union { v16bf v; v8bf hh[2]; } af;
        af.hh[0] = *(const v8bf*)&sC[wr * 16 + lrow][k0 + 16 * g];
        af.hh[1] = *(const v8bf*)&sC[wr * 16 + lrow][k0 + 16 * g + 8];
        #pragma unroll
        for (int ni = 0; ni < 2; ++ni) {
            v16bf bb = *(const v16bf*)&sB[(wc * 2 + ni) * 16 + lrow][k0 + 16 * g];
            accG[ni] = wmma_bf16(af.v, bb, accG[ni]);
        }
    }
    // ---- causal mask + decay/dt scaling -> G (bf16, LDS) ----
    #pragma unroll
    for (int ni = 0; ni < 2; ++ni) {
        const int j = (wc * 2 + ni) * 16 + lrow;
        union { v8f v; float f[8]; } res; res.v = accG[ni];
        #pragma unroll
        for (int e = 0; e < 8; ++e) {
            const int i = wr * 16 + g * 8 + e;
            const float v = (i >= j)
                ? __expf(sLa[i] - sLa[j]) * sdt[j] * res.f[e] : 0.f;
            sG[i][j] = to_bf16(v);
        }
    }
    __syncthreads();

    // ---- GEMM2: Y1 = G * X (K = 64); GEMM3: Y2 = C * Hprev (K = 128) ----
    v8f acc1[2], acc3[2];
    acc1[0] = (v8f)(0.f); acc1[1] = (v8f)(0.f);
    acc3[0] = (v8f)(0.f); acc3[1] = (v8f)(0.f);
    #pragma unroll
    for (int k0 = 0; k0 < CHQ; k0 += 32) {
        union { v16bf v; v8bf hh[2]; } af;
        af.hh[0] = *(const v8bf*)&sG[wr * 16 + lrow][k0 + 16 * g];
        af.hh[1] = *(const v8bf*)&sG[wr * 16 + lrow][k0 + 16 * g + 8];
        #pragma unroll
        for (int ni = 0; ni < 2; ++ni) {
            v16bf bb = *(const v16bf*)&sXt[(wc * 2 + ni) * 16 + lrow][k0 + 16 * g];
            acc1[ni] = wmma_bf16(af.v, bb, acc1[ni]);
        }
    }
    #pragma unroll
    for (int k0 = 0; k0 < DSTATE; k0 += 32) {
        union { v16bf v; v8bf hh[2]; } af;
        af.hh[0] = *(const v8bf*)&sC[wr * 16 + lrow][k0 + 16 * g];
        af.hh[1] = *(const v8bf*)&sC[wr * 16 + lrow][k0 + 16 * g + 8];
        #pragma unroll
        for (int ni = 0; ni < 2; ++ni) {
            v16bf bb = *(const v16bf*)&sHt[(wc * 2 + ni) * 16 + lrow][k0 + 16 * g];
            acc3[ni] = wmma_bf16(af.v, bb, acc3[ni]);
        }
    }

    // ---- combine + store ----
    const float Dh = Dp[h];
    #pragma unroll
    for (int ni = 0; ni < 2; ++ni) {
        const int p = (wc * 2 + ni) * 16 + lrow;
        union { v8f v; float f[8]; } r1, r3;
        r1.v = acc1[ni]; r3.v = acc3[ni];
        #pragma unroll
        for (int e = 0; e < 8; ++e) {
            const int i = wr * 16 + g * 8 + e;
            const float cum = __expf(sLa[i]);
            const float xv  = bf16_to_f32(sXt[p][i]);
            y[(rowBase + i) * DI + h * HD + p] = r1.f[e] + cum * r3.f[e] + Dh * xv;
        }
    }
}

// ---------------------------------------------------------------------------
// y *= silu(z); RMSNorm over d_inner; emit bf16 for the out-proj GEMM
// ---------------------------------------------------------------------------
__global__ __launch_bounds__(256) void k_gate_norm(float* __restrict__ y,
                                                   const float* __restrict__ zx,
                                                   const float* __restrict__ nw,
                                                   bf16* __restrict__ ybf) {
    __shared__ float red[256];
    const int r = blockIdx.x;
    const int t = threadIdx.x;
    const float* zrow = zx + (size_t)r * DPROJ;
    float*       yrow = y  + (size_t)r * DI;

    float ss = 0.f;
    #pragma unroll
    for (int i = t; i < DI; i += 256) {
        const float v = yrow[i] * silu(zrow[i]);
        yrow[i] = v;
        ss += v * v;
    }
    red[t] = ss;
    __syncthreads();
    for (int s = 128; s > 0; s >>= 1) {
        if (t < s) red[t] += red[t + s];
        __syncthreads();
    }
    const float rms = rsqrtf(red[0] / (float)DI + 1e-5f);

    bf16* orow = ybf + (size_t)r * DI;
    #pragma unroll
    for (int i = t; i < DI; i += 256)
        orow[i] = to_bf16(yrow[i] * rms * nw[i]);
}

// ---------------------------------------------------------------------------
// Launch
// ---------------------------------------------------------------------------
extern "C" void kernel_launch(void* const* d_in, const int* in_sizes, int n_in,
                              void* d_out, int out_size, void* d_ws, size_t ws_size,
                              hipStream_t stream) {
    const float* x       = (const float*)d_in[0];
    const float* W_in    = (const float*)d_in[1];
    const float* conv_w  = (const float*)d_in[2];
    const float* conv_b  = (const float*)d_in[3];
    const float* dt_bias = (const float*)d_in[4];
    const float* A_log   = (const float*)d_in[5];
    const float* Dp      = (const float*)d_in[6];
    const float* norm_w  = (const float*)d_in[7];
    const float* W_out   = (const float*)d_in[8];
    float* out = (float*)d_out;

    (void)in_sizes; (void)n_in; (void)out_size; (void)ws_size;

    char* ws = (char*)d_ws;
    size_t off = 0;
    auto take = [&](size_t bytes) -> char* {
        char* p = ws + off;
        off = (off + bytes + 255) & ~(size_t)255;
        return p;
    };
    bf16*  Xbf    = (bf16*)take((size_t)MM * DM * sizeof(bf16));
    bf16*  Wibf   = (bf16*)take((size_t)DPROJ * DM * sizeof(bf16));
    bf16*  Wobf   = (bf16*)take((size_t)DM * DI * sizeof(bf16));
    float* zx     = (float*)take((size_t)MM * DPROJ * sizeof(float));
    float* xBCc   = (float*)take((size_t)MM * CDIM * sizeof(float));
    float* dtv    = (float*)take((size_t)MM * NH * sizeof(float));
    float* logav  = (float*)take((size_t)MM * NH * sizeof(float));
    float* ybuf   = (float*)take((size_t)MM * DI * sizeof(float));
    bf16*  ybf    = (bf16*)take((size_t)MM * DI * sizeof(bf16));
    float* chunkS = (float*)take((size_t)NBHC * DSTATE * HD * sizeof(float)); // 268 MB
    float* chunkH = (float*)take((size_t)NBHC * DSTATE * HD * sizeof(float)); // 268 MB
    float* chunkA = (float*)take((size_t)NBHC * sizeof(float));

    // 1) fp32 -> bf16 conversions
    {
        size_t n = (size_t)MM * DM;
        k_f32_to_bf16<<<dim3((unsigned)((n + 255) / 256)), dim3(256), 0, stream>>>(x, Xbf, n);
        n = (size_t)DPROJ * DM;
        k_f32_to_bf16<<<dim3((unsigned)((n + 255) / 256)), dim3(256), 0, stream>>>(W_in, Wibf, n);
        n = (size_t)DM * DI;
        k_f32_to_bf16<<<dim3((unsigned)((n + 255) / 256)), dim3(256), 0, stream>>>(W_out, Wobf, n);
    }

    // 2) in-proj GEMM
    {
        dim3 grid((DPROJ + 127) / 128, MM / 128);
        k_gemm_bf16_wmma<<<grid, dim3(256), 0, stream>>>(Xbf, Wibf, zx, MM, DPROJ, DM);
    }

    // 3) causal conv + SiLU
    {
        size_t n = (size_t)MM * CDIM;
        k_conv_silu<<<dim3((unsigned)((n + 255) / 256)), dim3(256), 0, stream>>>(zx, conv_w, conv_b, xBCc);
    }

    // 4) dt / log-decay
    {
        size_t n = (size_t)MM * NH;
        k_dt_decay<<<dim3((unsigned)((n + 255) / 256)), dim3(256), 0, stream>>>(zx, dt_bias, A_log, dtv, logav);
    }

    // 5) chunked scan: states -> short recurrence -> outputs
    k_chunk_state<<<dim3(NBHC), dim3(256), 0, stream>>>(xBCc, dtv, logav, chunkS, chunkA);
    k_chunk_scan<<<dim3(BB * NH), dim3(256), 0, stream>>>(chunkS, chunkA, chunkH);
    k_chunk_out<<<dim3(NBHC), dim3(256), 0, stream>>>(xBCc, dtv, logav, chunkH, Dp, ybuf);

    // 6) gate + RMSNorm -> bf16
    k_gate_norm<<<dim3(MM), dim3(256), 0, stream>>>(ybuf, zx, norm_w, ybf);

    // 7) out-proj GEMM
    {
        dim3 grid((DM + 127) / 128, MM / 128);
        k_gemm_bf16_wmma<<<grid, dim3(256), 0, stream>>>(ybf, Wobf, out, MM, DM, DI);
    }
}
